// EdgeDecoder_15453292331281
// MI455X (gfx1250) — matrix operand order repaired and checked by reading, not compiled, not run on hardware
//
#include <hip/hip_runtime.h>
#include <hip/hip_bf16.h>

// ---------------------------------------------------------------------------
// EdgeDecoder: out[e] = relu(concat(z[row[e]], z[col[e]]) @ W1 + b1) @ W2 + b2
// bf16 WMMA (v_wmma_f32_16x16x32_bf16) with f32 accumulate.
//  - z pre-converted to bf16 in d_ws; W1 transposed+converted to bf16.
//  - Gather: wave-uniform edge index via SMEM, per-lane 16B copied straight
//    into LDS with global_load_async_to_lds_b128 (ASYNCcnt path).
//  - N-reduction via DPP16 row_shr cascade (no LDS permutes).
// ---------------------------------------------------------------------------

typedef __attribute__((ext_vector_type(16))) __bf16 bf16x16;
typedef __attribute__((ext_vector_type(8)))  float  f32x8;

__device__ __forceinline__ unsigned short f2bf(float f) {
  unsigned int u = __float_as_uint(f);
  unsigned int r = u + 0x7FFFu + ((u >> 16) & 1u);   // round-to-nearest-even
  return (unsigned short)(r >> 16);
}

union Frag32B {            // one WMMA 16-bit operand fragment (32B per lane)
  bf16x16 v;
  uint4   q[2];
};

// DPP16 row_shr:n accumulate; CTRL must be a compile-time constant.
template <int CTRL>
__device__ __forceinline__ float dpp_add(float x) {
  int sh = __builtin_amdgcn_update_dpp(0, __float_as_int(x), CTRL,
                                       0xf, 0xf, /*bound_ctrl=*/true);
  return x + __int_as_float(sh);
}

// --- prep: W1 [256,128] f32 row-major -> W1^T [128,256] bf16 (n-major) ------
__global__ void prep_w1t_kernel(const float* __restrict__ W1,
                                unsigned short* __restrict__ w1t) {
  int i = blockIdx.x * 256 + threadIdx.x;        // 0 .. 32767
  if (i < 128 * 256) {
    int n = i >> 8;          // 0..127
    int k = i & 255;         // 0..255
    w1t[i] = f2bf(W1[k * 128 + n]);
  }
}

// --- prep: z f32 -> bf16, 8 elements per thread -----------------------------
__global__ void prep_z_kernel(const float* __restrict__ z,
                              unsigned short* __restrict__ zb, int n8) {
  int i = blockIdx.x * blockDim.x + threadIdx.x;
  if (i < n8) {
    const float4* p = (const float4*)(z + (size_t)i * 8);
    float4 a = p[0], b = p[1];
    uint4 o;
    o.x = (unsigned)f2bf(a.x) | ((unsigned)f2bf(a.y) << 16);
    o.y = (unsigned)f2bf(a.z) | ((unsigned)f2bf(a.w) << 16);
    o.z = (unsigned)f2bf(b.x) | ((unsigned)f2bf(b.y) << 16);
    o.w = (unsigned)f2bf(b.z) | ((unsigned)f2bf(b.w) << 16);
    *(uint4*)(zb + (size_t)i * 8) = o;
  }
}

// --- main kernel ------------------------------------------------------------
// Block = 256 threads = 8 wave32. Wave w owns hidden dims [16w, 16w+16).
// Block processes 256 edges as 16 M-tiles of 16 edges.
// Gather: wave w moves edge (eb+w) [it=0] and (eb+w+8) [it=1]; 32 lanes x 16B
// = the edge's full 512B bf16 concat row (lanes 0-15: z[row], 16-31: z[col]).
template <bool ZB>
__global__ __launch_bounds__(256)
void edge_decoder_kernel(const float* __restrict__ z,
                         const unsigned short* __restrict__ zb,
                         const int* __restrict__ eidx,
                         const unsigned short* __restrict__ w1t,
                         const float* __restrict__ b1,
                         const float* __restrict__ W2,
                         const float* __restrict__ b2,
                         float* __restrict__ out, int E) {
  constexpr int AT_STRIDE = 264;                       // 256 + 8 pad (bank-safe)
  __shared__ __align__(16) unsigned short a_tile[16 * AT_STRIDE]; // 8448 B
  __shared__ float red2[8][16];

  const int t    = threadIdx.x;
  const int lane = t & 31;
  const int wv   = __builtin_amdgcn_readfirstlane(t >> 5);  // uniform wave id
  const int nloc = lane & 15;         // N within tile (C/D layout: N = lane%16)
  const int khi  = lane >> 4;         // 0 or 1 (lane half)

  // Per-wave constants for this N column group.
  const float bias1 = b1[(wv << 4) + nloc];
  const float scal2 = W2[(wv << 4) + nloc];
  const float bias2 = b2[0];

  // Load B fragments (W1^T slice for this wave): 8 K-chunks, resident in VGPRs.
  // B 32x16 operand, contiguous k order chosen here: lane = N + 16*khalf, idx j <-> K = 16*khalf + j.
  Frag32B bfrag[8];
  {
    const uint4* bp = (const uint4*)(w1t + ((wv << 4) + nloc) * 256 + (khi << 4));
#pragma unroll
    for (int kc = 0; kc < 8; ++kc) {
      bfrag[kc].q[0] = bp[kc * 4 + 0];
      bfrag[kc].q[1] = bp[kc * 4 + 1];
    }
  }

  const int ebBlock = blockIdx.x * 256;

  for (int mt = 0; mt < 16; ++mt) {
    const int eb = ebBlock + (mt << 4);

    // ---- gather stage: 2 edges per wave, 16B per lane, branch-free ---------
#pragma unroll
    for (int it = 0; it < 2; ++it) {
      const int m = wv + (it << 3);            // row in tile (uniform)
      int e = eb + m;
      e = (e < E) ? e : 0;                     // uniform clamp (scalar select)
      const int nrow = eidx[e];                // wave-uniform -> s_load
      const int ncol = eidx[E + e];            // wave-uniform -> s_load
      const int node = khi ? ncol : nrow;      // per lane-half
      const int off  = nloc << 3;              // feature chunk within endpoint
      if (ZB) {
        const unsigned short* gp = zb + (((size_t)node) << 7) + off;
        unsigned lds_off =
            (unsigned)(size_t)&a_tile[m * AT_STRIDE + (lane << 3)];
        asm volatile("global_load_async_to_lds_b128 %0, %1, off"
                     :: "v"(lds_off), "v"(gp)
                     : "memory");
      } else {
        const float4* zp = (const float4*)(z + (((size_t)node) << 7) + off);
        float4 a = zp[0], b = zp[1];
        uint4 pk;
        pk.x = (unsigned)f2bf(a.x) | ((unsigned)f2bf(a.y) << 16);
        pk.y = (unsigned)f2bf(a.z) | ((unsigned)f2bf(a.w) << 16);
        pk.z = (unsigned)f2bf(b.x) | ((unsigned)f2bf(b.y) << 16);
        pk.w = (unsigned)f2bf(b.z) | ((unsigned)f2bf(b.w) << 16);
        *(uint4*)&a_tile[m * AT_STRIDE + (lane << 3)] = pk;
      }
    }
    if (ZB) {
      asm volatile("s_wait_asynccnt 0x0" ::: "memory");
    }
    __syncthreads();

    // ---- WMMA stage: acc(16x16) = A(16x256) @ W1tile(256x16) + b1 ----------
    f32x8 acc;
#pragma unroll
    for (int i = 0; i < 8; ++i) acc[i] = bias1;

    // A 16x32 operand, contiguous k order chosen here: lane = M + 16*khalf;
    //   idx j <-> K = 16*khalf + j (matches B)
    const unsigned short* arow = &a_tile[nloc * AT_STRIDE]; // M = lane%16
    const int kh8 = khi << 4;
#pragma unroll
    for (int kc = 0; kc < 8; ++kc) {
      Frag32B af;
      const int K0 = kc << 5;
      af.q[0] = *(const uint4*)(arow + K0 + kh8);
      af.q[1] = *(const uint4*)(arow + K0 + kh8 + 8);
      acc = __builtin_amdgcn_wmma_f32_16x16x32_bf16(
          false, af.v, false, bfrag[kc].v, (short)0, acc, false, false);
    }

    // ---- ReLU, scale by W2[n], DPP 16-lane reduction -----------------------
    // C/D layout: lanes 0-15: N=lane, M=v; lanes 16-31: N=lane-16, M=v+8.
    float p[8];
#pragma unroll
    for (int v = 0; v < 8; ++v) {
      float h = acc[v] > 0.f ? acc[v] : 0.f;
      p[v] = h * scal2;
    }
#pragma unroll
    for (int v = 0; v < 8; ++v) {
      p[v] = dpp_add<0x111>(p[v]);   // row_shr:1
      p[v] = dpp_add<0x112>(p[v]);   // row_shr:2
      p[v] = dpp_add<0x114>(p[v]);   // row_shr:4
      p[v] = dpp_add<0x118>(p[v]);   // row_shr:8 -> lane15/31 have row sums
    }
    if (nloc == 15) {                       // lanes 15 and 31
      const int half = khi << 3;            // rows 0..7 or 8..15
#pragma unroll
      for (int v = 0; v < 8; ++v) red2[wv][half + v] = p[v];
    }
    __syncthreads();

    // ---- combine 8 waves' partials, add b2, store --------------------------
    if (t < 16) {
      int e = eb + t;
      if (e < E) {
        float s = bias2;
#pragma unroll
        for (int w = 0; w < 8; ++w) s += red2[w][t];
        out[e] = s;
      }
    }
    __syncthreads();
  }
}

// ---------------------------------------------------------------------------
extern "C" void kernel_launch(void* const* d_in, const int* in_sizes, int n_in,
                              void* d_out, int out_size, void* d_ws, size_t ws_size,
                              hipStream_t stream) {
  const float* z    = (const float*)d_in[0];
  const int*   eidx = (const int*)d_in[1];
  const float* W1   = (const float*)d_in[2];
  const float* b1   = (const float*)d_in[3];
  const float* W2   = (const float*)d_in[4];
  const float* b2   = (const float*)d_in[5];
  float* out = (float*)d_out;

  const int NZ = in_sizes[0];          // node-embedding float count (N*128)
  const int E  = in_sizes[1] / 2;      // edge count

  const size_t zbytes = (size_t)NZ * 2;
  const size_t zpad   = (zbytes + 255) & ~(size_t)255;
  const bool   zb_ok  = ws_size >= zpad + 65536;

  unsigned short* ws_z   = (unsigned short*)d_ws;
  unsigned short* ws_w1t = zb_ok ? (unsigned short*)((char*)d_ws + zpad)
                                 : (unsigned short*)d_ws;

  // Prep: W1 transpose+convert (64KB), optional z convert (25.6MB).
  prep_w1t_kernel<<<128, 256, 0, stream>>>(W1, ws_w1t);
  if (zb_ok) {
    int n8 = NZ / 8;
    prep_z_kernel<<<(n8 + 255) / 256, 256, 0, stream>>>(z, ws_z, n8);
  }

  const int blocks = (E + 255) / 256;
  if (zb_ok) {
    edge_decoder_kernel<true><<<blocks, 256, 0, stream>>>(
        z, ws_z, eidx, ws_w1t, b1, W2, b2, out, E);
  } else {
    edge_decoder_kernel<false><<<blocks, 256, 0, stream>>>(
        z, ws_z, eidx, ws_w1t, b1, W2, b2, out, E);
  }
}
